// Attention_81372450390110
// MI455X (gfx1250) — compile-verified, matrix operand order checked by reading, not run on hardware
//
#include <hip/hip_runtime.h>

// Problem constants (match reference)
#define B_    2
#define S_    2048
#define D_    2048
#define NH_   16
#define KVH_  4
#define HD_   128
#define NQ_   (NH_ * HD_)          // 2048
#define NKV_  (KVH_ * HD_)         // 512
#define NCAT_ (NQ_ + 2 * NKV_)     // 3072
#define ROWS_ (B_ * S_)            // 4096
#define THETA_ 10000.0f
#define EPS_   1e-6f

typedef __bf16 bf16;
typedef __attribute__((ext_vector_type(16))) __bf16 v16bf;
typedef __attribute__((ext_vector_type(8)))  float  v8f;

union FragU { uint4 u[2]; v16bf f; };

// 16-bit A/B fragment loader for v_wmma_*_16x16x32_bf16.
// Matrix stored row-major [row][K], ld = row stride in elements.
// Per ISA layout: lanes 0-15 hold K = {k0..k0+7, k0+16..k0+23},
// lanes 16-31 hold K = {k0+8..k0+15, k0+24..k0+31}; row = (lane&15) offset.
__device__ __forceinline__ v16bf load_frag16(const bf16* base, int ld,
                                             int row, int k0, int lane) {
  const bf16* p = base + (size_t)row * ld + k0 + ((lane >> 4) << 3);
  FragU fr;
  fr.u[0] = *reinterpret_cast<const uint4*>(p);
  fr.u[1] = *reinterpret_cast<const uint4*>(p + 16);
  return fr.f;
}

__device__ __forceinline__ v8f wmma_bf16(v16bf a, v16bf b, v8f c) {
  // (neg_a, A, neg_b, B, c_mod, C, reuse_a, reuse_b)
  return __builtin_amdgcn_wmma_f32_16x16x32_bf16(false, a, false, b,
                                                 (short)0, c, false, false);
}

// Async global->LDS copy (16 bytes per lane), tracked by ASYNCcnt.
// LDS offset = low 32 bits of the generic address (aperture rule).
__device__ __forceinline__ void async_ld_b128(bf16* lds_ptr, const bf16* gptr) {
  unsigned loff = (unsigned)(uintptr_t)lds_ptr;
  asm volatile("global_load_async_to_lds_b128 %0, %1, off"
               :: "v"(loff), "v"(gptr) : "memory");
}
__device__ __forceinline__ void wait_async0() {
  asm volatile("s_wait_asynccnt 0x0" ::: "memory");
}

// C/D 16x16 f32 tile: vgpr r holds M = r (lanes 0-15) / M = r+8 (lanes 16-31),
// N = lane & 15.
__device__ __forceinline__ void store_tile_bf16(bf16* out, int ld,
                                                int m0, int n0, v8f c, int lane) {
  int n  = n0 + (lane & 15);
  int mb = m0 + ((lane >> 4) << 3);
#pragma unroll
  for (int r = 0; r < 8; ++r) out[(size_t)(mb + r) * ld + n] = (bf16)c[r];
}

__device__ __forceinline__ void store_tile_f32(float* out, int ld,
                                               int m0, int n0, v8f c, int lane) {
  int n  = n0 + (lane & 15);
  int mb = m0 + ((lane >> 4) << 3);
#pragma unroll
  for (int r = 0; r < 8; ++r) out[(size_t)(mb + r) * ld + n] = c[r];
}

// ---------------------------------------------------------------- stage 0 ---
__global__ void k_cvt_x(const float* __restrict__ x, bf16* __restrict__ xb) {
  size_t i = (size_t)blockIdx.x * blockDim.x + threadIdx.x;
  if (i < (size_t)ROWS_ * D_) xb[i] = (bf16)x[i];
}

// Pack wq|wk|wv transposed into wcatT [NCAT_][D_], and wo transposed into
// woT [D_][NQ_] (N-major, K contiguous -> 16B contiguous B-fragment loads).
__global__ void k_pack_w(const float* __restrict__ wq, const float* __restrict__ wk,
                         const float* __restrict__ wv, const float* __restrict__ wo,
                         bf16* __restrict__ wcatT, bf16* __restrict__ woT) {
  size_t i  = (size_t)blockIdx.x * blockDim.x + threadIdx.x;
  size_t t1 = (size_t)NCAT_ * D_;
  if (i < t1) {
    int n = (int)(i / D_), k = (int)(i % D_);
    float v;
    if (n < NQ_)              v = wq[(size_t)k * NQ_ + n];
    else if (n < NQ_ + NKV_)  v = wk[(size_t)k * NKV_ + (n - NQ_)];
    else                      v = wv[(size_t)k * NKV_ + (n - NQ_ - NKV_)];
    wcatT[i] = (bf16)v;
  } else {
    size_t j = i - t1;
    if (j < (size_t)D_ * NQ_) {
      int n = (int)(j / NQ_), k = (int)(j % NQ_);
      woT[j] = (bf16)wo[(size_t)k * D_ + n];
    }
  }
}

// -------------------------------------------------- cooperative GEMM core ---
// C block 64(m) x 128(n) per 256-thread workgroup; 8 waves as 2(m) x 4(n),
// each wave a 32x32 macro-tile (2x2 WMMA). K panels (A: 64x32, B: 128x32)
// are double-buffered in LDS, filled with global_load_async_to_lds_b128 and
// consumed via ds_load_b128 fragment reads. Barriers are uniform (exact grid).
template <int LDA_, int LDB_, int KDIM_>
__device__ __forceinline__ void gemm_block_staged(
    const bf16* __restrict__ Ag, const bf16* __restrict__ Bg,
    int m0, int n0, bf16* As, bf16* Bs,   // As: [2][64*32], Bs: [2][128*32]
    v8f* acc /*[4]: c00,c01,c10,c11*/) {
  const int tid  = threadIdx.x;
  const int lane = tid & 31;
  const int wave = tid >> 5;
  const int wm = wave & 1, wn = wave >> 1;

  // Per-thread async-copy assignments (16B each).
  const int arow = tid >> 2, ach = tid & 3;                 // A: 256 xfers
  const bf16* agp = Ag + (size_t)(m0 + arow) * LDA_ + ach * 8;
  bf16* alp = As + arow * 32 + ach * 8;
  const int brow0 = (tid * 2) >> 2,     bch0 = (tid * 2) & 3;      // B: 512 xfers
  const int brow1 = (tid * 2 + 1) >> 2, bch1 = (tid * 2 + 1) & 3;
  const bf16* bgp0 = Bg + (size_t)(n0 + brow0) * LDB_ + bch0 * 8;
  const bf16* bgp1 = Bg + (size_t)(n0 + brow1) * LDB_ + bch1 * 8;
  bf16* blp0 = Bs + brow0 * 32 + bch0 * 8;
  bf16* blp1 = Bs + brow1 * 32 + bch1 * 8;

  auto issue = [&](int kk, int buf) {
    async_ld_b128(alp  + buf * (64 * 32),  agp  + kk);
    async_ld_b128(blp0 + buf * (128 * 32), bgp0 + kk);
    async_ld_b128(blp1 + buf * (128 * 32), bgp1 + kk);
  };

  // Wave fragment rows inside the LDS panels.
  const int rA0 = wm * 32 + (lane & 15), rA1 = rA0 + 16;
  const int rB0 = wn * 32 + (lane & 15), rB1 = rB0 + 16;

  int buf = 0;
  issue(0, 0);
  wait_async0();
  __syncthreads();

  for (int k0 = 0; k0 < KDIM_; k0 += 32) {
    if (k0 + 32 < KDIM_) issue(k0 + 32, buf ^ 1);

    const bf16* ap = As + buf * (64 * 32);
    const bf16* bp = Bs + buf * (128 * 32);
    v16bf a0 = load_frag16(ap, 32, rA0, 0, lane);
    v16bf a1 = load_frag16(ap, 32, rA1, 0, lane);
    v16bf b0 = load_frag16(bp, 32, rB0, 0, lane);
    v16bf b1 = load_frag16(bp, 32, rB1, 0, lane);
    acc[0] = wmma_bf16(a0, b0, acc[0]);
    acc[1] = wmma_bf16(a0, b1, acc[1]);
    acc[2] = wmma_bf16(a1, b0, acc[2]);
    acc[3] = wmma_bf16(a1, b1, acc[3]);

    wait_async0();
    __syncthreads();
    buf ^= 1;
  }
}

// ---------------------------------------------------------------- stage 1 ---
// qkvb[ROWS_][NCAT_] = xb[ROWS_][D_] @ [wq|wk|wv]  (bf16 in, f32 acc, bf16 out)
__global__ void k_gemm_qkv(const bf16* __restrict__ xb,
                           const bf16* __restrict__ wt,
                           bf16* __restrict__ outb) {
  __shared__ __align__(16) bf16 As[2][64 * 32];    //  8 KB
  __shared__ __align__(16) bf16 Bs[2][128 * 32];   // 16 KB
  const int tiles_n = NCAT_ / 128;                 // 24
  int bm = blockIdx.x / tiles_n, bn = blockIdx.x % tiles_n;
  int m0 = bm * 64, n0 = bn * 128;

  v8f z = {};
  v8f acc[4] = {z, z, z, z};
  gemm_block_staged<D_, D_, D_>(xb, wt, m0, n0, &As[0][0], &Bs[0][0], acc);

  const int lane = threadIdx.x & 31;
  const int wave = threadIdx.x >> 5;
  int wm0 = m0 + (wave & 1) * 32, wn0 = n0 + (wave >> 1) * 32;
  store_tile_bf16(outb, NCAT_, wm0,      wn0,      acc[0], lane);
  store_tile_bf16(outb, NCAT_, wm0,      wn0 + 16, acc[1], lane);
  store_tile_bf16(outb, NCAT_, wm0 + 16, wn0,      acc[2], lane);
  store_tile_bf16(outb, NCAT_, wm0 + 16, wn0 + 16, acc[3], lane);
}

// ---------------------------------------------------------------- stage 2 ---
// Per-head RMSNorm + RoPE for q/k; scatter to head-major bf16.
// v is scattered transposed: vbT[b][kvh][d][s] so PV B-fragments are contiguous.
// One wave handles one (b, head-slot, s) row of 128 elements (4 per lane).
__global__ void k_norm_rope(const bf16* __restrict__ qkv,
                            const float* __restrict__ qw, const float* __restrict__ kw,
                            bf16* __restrict__ qb, bf16* __restrict__ kb,
                            bf16* __restrict__ vbT) {
  const int lane = threadIdx.x & 31;
  int wrow = blockIdx.x * (blockDim.x >> 5) + (threadIdx.x >> 5);
  const int slots = NH_ + 2 * KVH_;                    // 24
  if (wrow >= B_ * slots * S_) return;
  int s  = wrow % S_;
  int t  = wrow / S_;
  int hh = t % slots;
  int b  = t / slots;

  int kind, h, col0;
  if (hh < NH_)            { kind = 0; h = hh;              col0 = h * HD_; }
  else if (hh < NH_ + KVH_){ kind = 1; h = hh - NH_;        col0 = NQ_ + h * HD_; }
  else                     { kind = 2; h = hh - NH_ - KVH_; col0 = NQ_ + NKV_ + h * HD_; }

  const bf16* rp = qkv + (size_t)(b * S_ + s) * NCAT_ + col0;
  int e0 = 4 * lane;
  float x0 = (float)rp[e0], x1 = (float)rp[e0 + 1];
  float x2 = (float)rp[e0 + 2], x3 = (float)rp[e0 + 3];

  if (kind == 2) {  // value: transpose-scatter, no norm/rope
    bf16* vp = vbT + ((size_t)(b * KVH_ + h) * HD_) * S_ + s;
    vp[(size_t)(e0 + 0) * S_] = (bf16)x0;
    vp[(size_t)(e0 + 1) * S_] = (bf16)x1;
    vp[(size_t)(e0 + 2) * S_] = (bf16)x2;
    vp[(size_t)(e0 + 3) * S_] = (bf16)x3;
    return;
  }

  // RMSNorm over 128 elements (full-wave shfl_xor reduction)
  float ss = x0 * x0 + x1 * x1 + x2 * x2 + x3 * x3;
  ss += __shfl_xor(ss, 16);
  ss += __shfl_xor(ss, 8);
  ss += __shfl_xor(ss, 4);
  ss += __shfl_xor(ss, 2);
  ss += __shfl_xor(ss, 1);
  float inv = rsqrtf(ss * (1.0f / HD_) + EPS_);
  const float* w = (kind == 0) ? qw : kw;
  x0 *= inv * w[e0];     x1 *= inv * w[e0 + 1];
  x2 *= inv * w[e0 + 2]; x3 *= inv * w[e0 + 3];

  // RoPE: pairs p0 = 2*lane (x0,x1), p1 = 2*lane+1 (x2,x3)
  float fs = (float)s;
  float a0 = fs * __powf(THETA_, -((float)(e0 + 0)) * (1.0f / HD_));
  float a1 = fs * __powf(THETA_, -((float)(e0 + 2)) * (1.0f / HD_));
  float c0 = __cosf(a0), s0 = __sinf(a0);
  float c1 = __cosf(a1), s1 = __sinf(a1);
  float y0 = x0 * c0 - x1 * s0, y1 = x0 * s0 + x1 * c0;
  float y2 = x2 * c1 - x3 * s1, y3 = x2 * s1 + x3 * c1;

  bf16* op = (kind == 0)
      ? qb + ((size_t)((b * NH_  + h) * S_ + s)) * HD_
      : kb + ((size_t)((b * KVH_ + h) * S_ + s)) * HD_;
  op[e0] = (bf16)y0; op[e0 + 1] = (bf16)y1;
  op[e0 + 2] = (bf16)y2; op[e0 + 3] = (bf16)y3;
}

// ---------------------------------------------------------------- stage 3 ---
// Flash attention: one wave per (b, h, 16-row q tile); stream 32-key blocks
// up to causal frontier; online softmax on WMMA C-layout; P re-fragmented
// via per-wave LDS tile; ctxb[b][s][h*HD+d] = O / l.
__global__ void k_attn(const bf16* __restrict__ qb, const bf16* __restrict__ kb,
                       const bf16* __restrict__ vbT, bf16* __restrict__ ctxb) {
  __shared__ __align__(16) bf16 p_lds[8][16 * 32];   // 1 KB per wave
  const int lane = threadIdx.x & 31;
  const int wave = threadIdx.x >> 5;
  int wt = blockIdx.x * 8 + wave;                    // B_*NH_*(S_/16) = 4096
  if (wt >= B_ * NH_ * (S_ / 16)) return;
  int qt = wt % (S_ / 16);
  int bh = wt / (S_ / 16);
  int h = bh % NH_, b = bh / NH_;
  int kvh = h >> 2;                                  // GQA: jnp.repeat -> h // 4
  int q0 = qt * 16;

  const bf16* qp = qb  + ((size_t)(b * NH_  + h)   * S_)  * HD_;
  const bf16* kp = kb  + ((size_t)(b * KVH_ + kvh) * S_)  * HD_;
  const bf16* vp = vbT + ((size_t)(b * KVH_ + kvh) * HD_) * S_;

  const int n_id   = lane & 15;
  const int m_base = (lane >> 4) << 3;

  // Persistent q fragments: 16 rows x 128 K
  v16bf qa[4];
  int rowA = q0 + n_id;
#pragma unroll
  for (int i = 0; i < 4; ++i) qa[i] = load_frag16(qp, HD_, rowA, i * 32, lane);

  v8f z = {};
  v8f o[8];
#pragma unroll
  for (int i = 0; i < 8; ++i) o[i] = z;
  float m_r[8], l_r[8];
#pragma unroll
  for (int r = 0; r < 8; ++r) { m_r[r] = -1e30f; l_r[r] = 0.0f; }

  const float scale = 0.08838834764831845f;          // 1/sqrt(128)
  const int kend = q0 + 16;                          // causal frontier (exclusive)

  for (int kb0 = 0; kb0 < kend; kb0 += 32) {
    // ---- scores: 16q x 32k = two C tiles, K = HD = 128
    v8f s0t = z, s1t = z;
    int rowB0 = kb0 + n_id, rowB1 = rowB0 + 16;
#pragma unroll
    for (int kk = 0; kk < 4; ++kk) {
      v16bf bk0 = load_frag16(kp, HD_, rowB0, kk * 32, lane);
      v16bf bk1 = load_frag16(kp, HD_, rowB1, kk * 32, lane);
      s0t = wmma_bf16(qa[kk], bk0, s0t);
      s1t = wmma_bf16(qa[kk], bk1, s1t);
    }

    // ---- scale + causal mask + online softmax (rows live in half-waves)
    float alpha[8];
#pragma unroll
    for (int r = 0; r < 8; ++r) {
      int qi = q0 + m_base + r;
      float v0 = s0t[r] * scale; if (kb0 + n_id      > qi) v0 = -1e30f;
      float v1 = s1t[r] * scale; if (kb0 + 16 + n_id > qi) v1 = -1e30f;
      s0t[r] = v0; s1t[r] = v1;
      float mx = fmaxf(v0, v1);
      mx = fmaxf(mx, __shfl_xor(mx, 1));
      mx = fmaxf(mx, __shfl_xor(mx, 2));
      mx = fmaxf(mx, __shfl_xor(mx, 4));
      mx = fmaxf(mx, __shfl_xor(mx, 8));
      float mn = fmaxf(m_r[r], mx);
      alpha[r] = __expf(m_r[r] - mn);
      m_r[r] = mn;
      float p0 = __expf(v0 - mn), p1 = __expf(v1 - mn);
      s0t[r] = p0; s1t[r] = p1;
      float rs = p0 + p1;
      rs += __shfl_xor(rs, 1);
      rs += __shfl_xor(rs, 2);
      rs += __shfl_xor(rs, 4);
      rs += __shfl_xor(rs, 8);
      l_r[r] = l_r[r] * alpha[r] + rs;
    }
#pragma unroll
    for (int i = 0; i < 8; ++i)
#pragma unroll
      for (int r = 0; r < 8; ++r) o[i][r] *= alpha[r];

    // ---- re-fragment P through LDS (C layout -> A layout)
    bf16* pl = p_lds[wave];
#pragma unroll
    for (int r = 0; r < 8; ++r) {
      pl[(m_base + r) * 32 + n_id]      = (bf16)s0t[r];
      pl[(m_base + r) * 32 + 16 + n_id] = (bf16)s1t[r];
    }
    v16bf pa = load_frag16(pl, 32, n_id, 0, lane);

    // ---- O += P @ V   (V transposed layout: [d][s], K = 32 keys)
#pragma unroll
    for (int dt = 0; dt < 8; ++dt) {
      v16bf vb = load_frag16(vp, S_, dt * 16 + n_id, kb0, lane);
      o[dt] = wmma_bf16(pa, vb, o[dt]);
    }
  }

  // ---- epilogue: normalize by l and scatter to ctxb [b][s][h*HD+d]
  bf16* op = ctxb + ((size_t)b * S_) * NQ_ + h * HD_;
#pragma unroll
  for (int dt = 0; dt < 8; ++dt) {
#pragma unroll
    for (int r = 0; r < 8; ++r) {
      int srow = q0 + m_base + r;
      op[(size_t)srow * NQ_ + dt * 16 + n_id] = (bf16)(o[dt][r] / l_r[r]);
    }
  }
}

// ---------------------------------------------------------------- stage 4 ---
// out[ROWS_][D_] (f32) = ctxb[ROWS_][NQ_] @ wo   (B from woT [D_][NQ_])
__global__ void k_gemm_out(const bf16* __restrict__ ctxb,
                           const bf16* __restrict__ woT,
                           float* __restrict__ out) {
  __shared__ __align__(16) bf16 As[2][64 * 32];    //  8 KB
  __shared__ __align__(16) bf16 Bs[2][128 * 32];   // 16 KB
  const int tiles_n = D_ / 128;                    // 16
  int bm = blockIdx.x / tiles_n, bn = blockIdx.x % tiles_n;
  int m0 = bm * 64, n0 = bn * 128;

  v8f z = {};
  v8f acc[4] = {z, z, z, z};
  gemm_block_staged<NQ_, NQ_, NQ_>(ctxb, woT, m0, n0, &As[0][0], &Bs[0][0], acc);

  const int lane = threadIdx.x & 31;
  const int wave = threadIdx.x >> 5;
  int wm0 = m0 + (wave & 1) * 32, wn0 = n0 + (wave >> 1) * 32;
  store_tile_f32(out, D_, wm0,      wn0,      acc[0], lane);
  store_tile_f32(out, D_, wm0,      wn0 + 16, acc[1], lane);
  store_tile_f32(out, D_, wm0 + 16, wn0,      acc[2], lane);
  store_tile_f32(out, D_, wm0 + 16, wn0 + 16, acc[3], lane);
}

// ------------------------------------------------------------------ launch --
extern "C" void kernel_launch(void* const* d_in, const int* in_sizes, int n_in,
                              void* d_out, int out_size, void* d_ws, size_t ws_size,
                              hipStream_t stream) {
  const float* x  = (const float*)d_in[0];
  const float* wq = (const float*)d_in[1];
  const float* wk = (const float*)d_in[2];
  const float* wv = (const float*)d_in[3];
  const float* wo = (const float*)d_in[4];
  const float* qw = (const float*)d_in[5];
  const float* kw = (const float*)d_in[6];
  float* out = (float*)d_out;

  // Workspace layout (bytes), ~100 MB total.
  char* ws = (char*)d_ws;
  size_t off = 0;
  bf16* xb    = (bf16*)(ws + off); off += (size_t)ROWS_ * D_    * 2;  // 16 MB
  bf16* wcatT = (bf16*)(ws + off); off += (size_t)NCAT_ * D_    * 2;  // 12 MB
  bf16* woT   = (bf16*)(ws + off); off += (size_t)D_    * NQ_   * 2;  //  8 MB
  bf16* qkvb  = (bf16*)(ws + off); off += (size_t)ROWS_ * NCAT_ * 2;  // 24 MB
  bf16* qbp   = (bf16*)(ws + off); off += (size_t)B_ * NH_  * S_ * HD_ * 2; // 16 MB
  bf16* kbp   = (bf16*)(ws + off); off += (size_t)B_ * KVH_ * S_ * HD_ * 2; //  4 MB
  bf16* vbT   = (bf16*)(ws + off); off += (size_t)B_ * KVH_ * HD_ * S_ * 2; //  4 MB
  bf16* ctxb  = (bf16*)(ws + off); off += (size_t)ROWS_ * NQ_   * 2;  // 16 MB
  (void)ws_size; (void)in_sizes; (void)n_in; (void)out_size;

  // Stage 0: precision conversion + weight packing (transposed bf16)
  {
    size_t n = (size_t)ROWS_ * D_;
    k_cvt_x<<<(unsigned)((n + 255) / 256), 256, 0, stream>>>(x, xb);
    size_t np = (size_t)NCAT_ * D_ + (size_t)D_ * NQ_;
    k_pack_w<<<(unsigned)((np + 255) / 256), 256, 0, stream>>>(wq, wk, wv, wo,
                                                               wcatT, woT);
  }
  // Stage 1: fused QKV projection (async-LDS staged WMMA, 64x128 blocks)
  {
    int blocks = (ROWS_ / 64) * (NCAT_ / 128);        // 64 * 24 = 1536
    k_gemm_qkv<<<blocks, 256, 0, stream>>>(xb, wcatT, qkvb);
  }
  // Stage 2: RMSNorm + RoPE + head-major scatter (v transposed)
  {
    int rows = B_ * (NH_ + 2 * KVH_) * S_;            // 98304 waves
    k_norm_rope<<<rows / 8, 256, 0, stream>>>(qkvb, qw, kw, qbp, kbp, vbT);
  }
  // Stage 3: causal flash attention (WMMA QK^T + PV, online softmax)
  {
    int waves = B_ * NH_ * (S_ / 16);                 // 4096
    k_attn<<<waves / 8, 256, 0, stream>>>(qbp, kbp, vbT, ctxb);
  }
  // Stage 4: output projection (async-LDS staged WMMA) -> fp32 d_out
  {
    int blocks = (ROWS_ / 64) * (D_ / 128);           // 64 * 16 = 1024
    k_gemm_out<<<blocks, 256, 0, stream>>>(ctxb, woT, out);
  }
}
